// AttnBlock_13237089206666
// MI455X (gfx1250) — matrix operand order repaired and checked by reading, not compiled, not run on hardware
//
#include <hip/hip_runtime.h>
#include <hip/hip_bf16.h>
#include <math.h>

typedef __attribute__((ext_vector_type(16))) _Float16 v16h;
typedef __attribute__((ext_vector_type(8)))  _Float16 v8h;
typedef __attribute__((ext_vector_type(8)))  float    v8f;

static constexpr int Bn  = 16;    // batch
static constexpr int Cc  = 512;   // channels
static constexpr int HWn = 1024;  // H*W
static constexpr int CG  = 16;    // channels per group (512/32)

// GEMM tiling: workgroup = 256 threads = 8 waves (wave32), wave grid 4x2,
// each wave computes 32x32 via 2x2 WMMA 16x16 accumulators.
static constexpr int MT = 128;
static constexpr int NT = 64;
static constexpr int KT = 64;

// ---- Async global->LDS staging availability --------------------------------
#if defined(__has_builtin)
#if __has_builtin(__builtin_amdgcn_global_load_async_to_lds_b128)
#define HAVE_ASYNC 1
#endif
#endif
#ifndef HAVE_ASYNC
#define HAVE_ASYNC 0
#endif

#if HAVE_ASYNC
typedef int vi4 __attribute__((vector_size(16)));
typedef __attribute__((address_space(1))) vi4* as1_vi4p;  // global 128-bit ptr
typedef __attribute__((address_space(3))) vi4* as3_vi4p;  // LDS    128-bit ptr
#endif

__device__ inline void async_wait_all() {
#if HAVE_ASYNC
  asm volatile("s_wait_asynccnt 0x0" ::: "memory");
#endif
}

__device__ inline v8f wmma_f16(v16h a, v16h b, v8f c) {
  return __builtin_amdgcn_wmma_f32_16x16x32_f16(false, a, false, b, (short)0, c,
                                                false, false);
}

// ---- LDS staging -----------------------------------------------------------
// dst is a [rows][KT] row-major f16 tile.
// F32 = false: f16 source, async DMA to LDS (16B per lane) when available.
// F32 = true : f32 source converted to f16 through VGPRs (weights).
template <bool F32>
__device__ inline void stage(_Float16* dst, const void* __restrict__ src,
                             int row0, int k0, int pitch, int rows, int tid) {
  if constexpr (F32) {
    const float* s = (const float*)src;
    const int total = rows * KT;
    for (int idx = tid; idx < total; idx += 256) {
      int r = idx >> 6;
      int c = idx & 63;
      dst[r * KT + c] = (_Float16)s[(size_t)(row0 + r) * pitch + k0 + c];
    }
  } else {
    const _Float16* s = (const _Float16*)src;
    const int chunks = rows * (KT / 8);  // 16-byte chunks
    for (int idx = tid; idx < chunks; idx += 256) {
      int r = idx >> 3;
      int c = (idx & 7) << 3;
      const _Float16* gp = s + (size_t)(row0 + r) * pitch + k0 + c;
      _Float16* lp = dst + r * KT + c;
#if HAVE_ASYNC
      __builtin_amdgcn_global_load_async_to_lds_b128(
          (as1_vi4p)(unsigned long long)gp,
          (as3_vi4p)(unsigned int)(unsigned long long)lp, 0, 0);
#else
      *(uint4*)lp = *(const uint4*)gp;
#endif
    }
  }
}

// ---- Fragment loads per CDNA5 ISA VGPR layouts -----------------------------
// A 16x32 f16: lanes 0-15 hold K {0..7,16..23}, lanes 16-31 hold K {8..15,24..31}.
__device__ inline v16h frag_a(const _Float16* As, int m16, int kk, int lane) {
  const _Float16* p = As + (size_t)(m16 + (lane & 15)) * KT + kk + ((lane >> 4) << 3);
  v8h lo = *(const v8h*)p;
  v8h hi = *(const v8h*)(p + 16);
  v16h f;
#pragma unroll
  for (int i = 0; i < 8; ++i) { f[i] = lo[i]; f[i + 8] = hi[i]; }
  return f;
}

// B 32x16 f16 (stored transposed [N][K] in LDS): lanes 0-15 hold K {0..7,16..23},
// lanes 16-31 hold K {8..15,24..31}; N = lane & 15.
__device__ inline v16h frag_b(const _Float16* Bs, int n16, int kk, int lane) {
  const _Float16* p = Bs + (size_t)(n16 + (lane & 15)) * KT + kk + ((lane >> 4) << 3);
  v8h lo = *(const v8h*)p;
  v8h hi = *(const v8h*)(p + 16);
  v16h f;
#pragma unroll
  for (int i = 0; i < 8; ++i) { f[i] = lo[i]; f[i + 8] = hi[i]; }
  return f;
}

__device__ inline void mma_tile(const _Float16* As, const _Float16* Bs, int wm,
                                int wn, int lane, v8f acc[2][2]) {
#pragma unroll
  for (int kk = 0; kk < KT; kk += 32) {
    v16h a0 = frag_a(As, wm * 32,      kk, lane);
    v16h a1 = frag_a(As, wm * 32 + 16, kk, lane);
    v16h b0 = frag_b(Bs, wn * 32,      kk, lane);
    v16h b1 = frag_b(Bs, wn * 32 + 16, kk, lane);
    acc[0][0] = wmma_f16(a0, b0, acc[0][0]);
    acc[0][1] = wmma_f16(a0, b1, acc[0][1]);
    acc[1][0] = wmma_f16(a1, b0, acc[1][0]);
    acc[1][1] = wmma_f16(a1, b1, acc[1][1]);
  }
}

// ---- Shared double-buffered GEMM main loop ---------------------------------
template <bool AF32, bool BF32>
__device__ inline void gemm_core(const void* __restrict__ Ap,
                                 const void* __restrict__ Bp, int m0, int n0,
                                 int apitch, int bpitch, int Ktot, int tid,
                                 v8f acc[2][2]) {
  __shared__ __align__(16) _Float16 As[2][MT * KT];  // 2 x 16 KB
  __shared__ __align__(16) _Float16 Bs[2][NT * KT];  // 2 x  8 KB
  const int lane = tid & 31, wid = tid >> 5;
  const int wm = wid >> 1, wn = wid & 1;

  stage<AF32>(As[0], Ap, m0, 0, apitch, MT, tid);
  stage<BF32>(Bs[0], Bp, n0, 0, bpitch, NT, tid);
  async_wait_all();
  __syncthreads();
  int cur = 0;
  for (int k0 = 0; k0 < Ktot; k0 += KT) {
    int nxt = cur ^ 1;
    if (k0 + KT < Ktot) {  // prefetch next K-tile while WMMAs run
      stage<AF32>(As[nxt], Ap, m0, k0 + KT, apitch, MT, tid);
      stage<BF32>(Bs[nxt], Bp, n0, k0 + KT, bpitch, NT, tid);
    }
    mma_tile(As[cur], Bs[cur], wm, wn, lane, acc);
    async_wait_all();
    __syncthreads();
    cur = nxt;
  }
}

// ---- GroupNorm -------------------------------------------------------------
__global__ __launch_bounds__(256) void gn_stats_kernel(const float* __restrict__ x,
                                                       float* __restrict__ stats) {
  const int bg = blockIdx.x;  // b*32 + g
  const float* p = x + (size_t)bg * (CG * HWn);
  const int tid = threadIdx.x;
  float s = 0.f, s2 = 0.f;
  for (int i = tid; i < CG * HWn; i += 256) {
    float t = p[i];
    s += t; s2 += t * t;
  }
  __shared__ float r1[256], r2[256];
  r1[tid] = s; r2[tid] = s2;
  __syncthreads();
  for (int st = 128; st > 0; st >>= 1) {
    if (tid < st) { r1[tid] += r1[tid + st]; r2[tid] += r2[tid + st]; }
    __syncthreads();
  }
  if (tid == 0) {
    const float inv = 1.0f / (CG * HWn);
    float mean = r1[0] * inv;
    float var  = r2[0] * inv - mean * mean;
    stats[bg * 2 + 0] = mean;
    stats[bg * 2 + 1] = rsqrtf(var + 1e-5f);
  }
}

// Normalize + transpose: hn[b][hw][c] (f16) from x[b][c][hw] (f32).
__global__ __launch_bounds__(256) void gn_apply_kernel(
    const float* __restrict__ x, const float* __restrict__ gw,
    const float* __restrict__ gb, const float* __restrict__ stats,
    _Float16* __restrict__ hn) {
  const int b = blockIdx.z;
  const int c0 = blockIdx.y * 32;
  const int h0 = blockIdx.x * 32;
  __shared__ float tile[32][33];
  const int tx = threadIdx.x & 31;
  const int ty = threadIdx.x >> 5;  // 8 rows
#pragma unroll
  for (int rr = 0; rr < 32; rr += 8) {
    int c = c0 + ty + rr;
    int g = c >> 4;
    float mu = stats[(b * 32 + g) * 2 + 0];
    float rs = stats[(b * 32 + g) * 2 + 1];
    float v = x[((size_t)b * Cc + c) * HWn + h0 + tx];
    tile[ty + rr][tx] = (v - mu) * rs * gw[c] + gb[c];
  }
  __syncthreads();
#pragma unroll
  for (int rr = 0; rr < 32; rr += 8) {
    int hw = h0 + ty + rr;
    hn[((size_t)b * HWn + hw) * Cc + c0 + tx] = (_Float16)tile[tx][ty + rr];
  }
}

// ---- QKV GEMM: out[m=hw][n=c_out] = hn[m][:]·W[n][:] + bias ----------------
// mode 0/1: store [HW][C] f16 (q scaled by C^-0.5); mode 2: store vT [C][HW].
__global__ __launch_bounds__(256) void qkv_kernel(
    const _Float16* __restrict__ hn, const float* __restrict__ w,
    const float* __restrict__ bias, _Float16* __restrict__ out, float scale,
    int mode) {
  const int b = blockIdx.z;
  const int m0 = blockIdx.x * MT;
  const int n0 = blockIdx.y * NT;
  const int tid = threadIdx.x, lane = tid & 31, wid = tid >> 5;
  const int wm = wid >> 1, wn = wid & 1;
  v8f acc[2][2] = {};
  gemm_core<false, true>(hn + (size_t)b * HWn * Cc, w, m0, n0, Cc, Cc, Cc, tid,
                         acc);
  const int mbase = m0 + wm * 32, nbase = n0 + wn * 32;
  const int nlane = lane & 15, mhi = (lane >> 4) * 8;
  if (mode < 2) {
    _Float16* dst = out + (size_t)b * HWn * Cc;
#pragma unroll
    for (int i = 0; i < 2; ++i)
#pragma unroll
      for (int j = 0; j < 2; ++j) {
        int n = nbase + j * 16 + nlane;
        float bv = bias[n];
#pragma unroll
        for (int r = 0; r < 8; ++r) {
          int m = mbase + i * 16 + mhi + r;
          dst[(size_t)m * Cc + n] = (_Float16)((acc[i][j][r] + bv) * scale);
        }
      }
  } else {  // vT[c][hw], packed 16-byte stores
    _Float16* dst = out + (size_t)b * Cc * HWn;
#pragma unroll
    for (int i = 0; i < 2; ++i)
#pragma unroll
      for (int j = 0; j < 2; ++j) {
        int n = nbase + j * 16 + nlane;
        float bv = bias[n];
        int m = mbase + i * 16 + mhi;
        v8h pk;
#pragma unroll
        for (int r = 0; r < 8; ++r) pk[r] = (_Float16)(acc[i][j][r] + bv);
        *(v8h*)(dst + (size_t)n * HWn + m) = pk;
      }
  }
}

// ---- Scores: S[p][P] = q[p][:]·k[P][:]  (scale already folded into q) ------
__global__ __launch_bounds__(256) void scores_kernel(
    const _Float16* __restrict__ q, const _Float16* __restrict__ kb,
    _Float16* __restrict__ s) {
  const int b = blockIdx.z;
  const int m0 = blockIdx.x * MT;
  const int n0 = blockIdx.y * NT;
  const int tid = threadIdx.x, lane = tid & 31, wid = tid >> 5;
  const int wm = wid >> 1, wn = wid & 1;
  v8f acc[2][2] = {};
  gemm_core<false, false>(q + (size_t)b * HWn * Cc, kb + (size_t)b * HWn * Cc,
                          m0, n0, Cc, Cc, Cc, tid, acc);
  _Float16* dst = s + (size_t)b * HWn * HWn;
  const int mbase = m0 + wm * 32, nbase = n0 + wn * 32;
  const int nlane = lane & 15, mhi = (lane >> 4) * 8;
#pragma unroll
  for (int i = 0; i < 2; ++i)
#pragma unroll
    for (int j = 0; j < 2; ++j) {
      int n = nbase + j * 16 + nlane;
#pragma unroll
      for (int r = 0; r < 8; ++r) {
        int m = mbase + i * 16 + mhi + r;
        dst[(size_t)m * HWn + n] = (_Float16)acc[i][j][r];
      }
    }
}

// ---- Row softmax over 1024 keys, f32 math, f16 in place --------------------
__global__ __launch_bounds__(256) void softmax_kernel(_Float16* __restrict__ s) {
  const size_t row = blockIdx.x;
  _Float16* p = s + row * HWn;
  const int tid = threadIdx.x;
  float v[4];
  float mx = -3.0e38f;
#pragma unroll
  for (int i = 0; i < 4; ++i) {
    v[i] = (float)p[tid + i * 256];
    mx = fmaxf(mx, v[i]);
  }
  __shared__ float red[256];
  red[tid] = mx;
  __syncthreads();
  for (int st = 128; st > 0; st >>= 1) {
    if (tid < st) red[tid] = fmaxf(red[tid], red[tid + st]);
    __syncthreads();
  }
  mx = red[0];
  __syncthreads();
  float sum = 0.f;
#pragma unroll
  for (int i = 0; i < 4; ++i) {
    v[i] = __expf(v[i] - mx);
    sum += v[i];
  }
  red[tid] = sum;
  __syncthreads();
  for (int st = 128; st > 0; st >>= 1) {
    if (tid < st) red[tid] += red[tid + st];
    __syncthreads();
  }
  float inv = 1.0f / red[0];
#pragma unroll
  for (int i = 0; i < 4; ++i) p[tid + i * 256] = (_Float16)(v[i] * inv);
}

// ---- O[p][c] = attn[p][:]·vT[c][:] -----------------------------------------
__global__ __launch_bounds__(256) void attnv_kernel(
    const _Float16* __restrict__ s, const _Float16* __restrict__ vT,
    _Float16* __restrict__ o) {
  const int b = blockIdx.z;
  const int m0 = blockIdx.x * MT;
  const int n0 = blockIdx.y * NT;
  const int tid = threadIdx.x, lane = tid & 31, wid = tid >> 5;
  const int wm = wid >> 1, wn = wid & 1;
  v8f acc[2][2] = {};
  gemm_core<false, false>(s + (size_t)b * HWn * HWn, vT + (size_t)b * Cc * HWn,
                          m0, n0, HWn, HWn, HWn, tid, acc);
  _Float16* dst = o + (size_t)b * HWn * Cc;
  const int mbase = m0 + wm * 32, nbase = n0 + wn * 32;
  const int nlane = lane & 15, mhi = (lane >> 4) * 8;
#pragma unroll
  for (int i = 0; i < 2; ++i)
#pragma unroll
    for (int j = 0; j < 2; ++j) {
      int n = nbase + j * 16 + nlane;
#pragma unroll
      for (int r = 0; r < 8; ++r) {
        int m = mbase + i * 16 + mhi + r;
        dst[(size_t)m * Cc + n] = (_Float16)acc[i][j][r];
      }
    }
}

// ---- out[d][hw] = p_w[d][:]·o[hw][:] + p_b[d] + x[d][hw]  (f32 out) --------
__global__ __launch_bounds__(256) void proj_kernel(
    const _Float16* __restrict__ o, const float* __restrict__ pw,
    const float* __restrict__ pb, const float* __restrict__ x,
    float* __restrict__ out) {
  const int b = blockIdx.z;
  const int m0 = blockIdx.x * MT;  // over channels d
  const int n0 = blockIdx.y * NT;  // over hw
  const int tid = threadIdx.x, lane = tid & 31, wid = tid >> 5;
  const int wm = wid >> 1, wn = wid & 1;
  v8f acc[2][2] = {};
  gemm_core<true, false>(pw, o + (size_t)b * HWn * Cc, m0, n0, Cc, Cc, Cc, tid,
                         acc);
  const float* xr = x + (size_t)b * Cc * HWn;
  float* dst = out + (size_t)b * Cc * HWn;
  const int mbase = m0 + wm * 32, nbase = n0 + wn * 32;
  const int nlane = lane & 15, mhi = (lane >> 4) * 8;
#pragma unroll
  for (int i = 0; i < 2; ++i)
#pragma unroll
    for (int j = 0; j < 2; ++j) {
      int n = nbase + j * 16 + nlane;
#pragma unroll
      for (int r = 0; r < 8; ++r) {
        int m = mbase + i * 16 + mhi + r;
        size_t idx = (size_t)m * HWn + n;
        dst[idx] = acc[i][j][r] + pb[m] + xr[idx];
      }
    }
}

extern "C" void kernel_launch(void* const* d_in, const int* in_sizes, int n_in,
                              void* d_out, int out_size, void* d_ws,
                              size_t ws_size, hipStream_t stream) {
  const float* x    = (const float*)d_in[0];
  // d_in[1] = temb (unused by reference)
  const float* gn_w = (const float*)d_in[2];
  const float* gn_b = (const float*)d_in[3];
  const float* q_w  = (const float*)d_in[4];
  const float* q_b  = (const float*)d_in[5];
  const float* k_w  = (const float*)d_in[6];
  const float* k_b  = (const float*)d_in[7];
  const float* v_w  = (const float*)d_in[8];
  const float* v_b  = (const float*)d_in[9];
  const float* p_w  = (const float*)d_in[10];
  const float* p_b  = (const float*)d_in[11];
  float* out = (float*)d_out;

  // Workspace carve-up (~96 MB): stats | hn(->o reuse) | q | k | vT | scores
  char* ws = (char*)d_ws;
  float* stats = (float*)ws;
  const size_t ACT = (size_t)Bn * HWn * Cc;  // 8M f16 elems = 16 MB
  _Float16* hn   = (_Float16*)(ws + 4096);
  _Float16* qbuf = hn + ACT;
  _Float16* kbuf = qbuf + ACT;
  _Float16* vT   = kbuf + ACT;
  _Float16* sc   = vT + ACT;                 // 16M elems = 32 MB
  _Float16* obuf = hn;                       // hn dead after QKV

  const float qscale = 1.0f / sqrtf((float)Cc);

  gn_stats_kernel<<<Bn * 32, 256, 0, stream>>>(x, stats);
  gn_apply_kernel<<<dim3(HWn / 32, Cc / 32, Bn), 256, 0, stream>>>(
      x, gn_w, gn_b, stats, hn);

  dim3 gqkv(HWn / MT, Cc / NT, Bn);  // 8 x 8 x 16
  qkv_kernel<<<gqkv, 256, 0, stream>>>(hn, q_w, q_b, qbuf, qscale, 0);
  qkv_kernel<<<gqkv, 256, 0, stream>>>(hn, k_w, k_b, kbuf, 1.0f, 1);
  qkv_kernel<<<gqkv, 256, 0, stream>>>(hn, v_w, v_b, vT, 1.0f, 2);

  scores_kernel<<<dim3(HWn / MT, HWn / NT, Bn), 256, 0, stream>>>(qbuf, kbuf, sc);
  softmax_kernel<<<Bn * HWn, 256, 0, stream>>>(sc);
  attnv_kernel<<<dim3(HWn / MT, Cc / NT, Bn), 256, 0, stream>>>(sc, vT, obuf);
  proj_kernel<<<dim3(Cc / MT, HWn / NT, Bn), 256, 0, stream>>>(obuf, p_w, p_b, x,
                                                               out);
}
